// NetFC1x1024DOcust_82660940578898
// MI455X (gfx1250) — compile-verified, matrix operand order checked by reading, not run on hardware
//
#include <hip/hip_runtime.h>
#include <stdint.h>
#include <stddef.h>

#define BB 8192
#define DD 3072
#define HH 2000
#define CC 1000

typedef __attribute__((ext_vector_type(16))) __bf16 v16bf;
typedef __attribute__((ext_vector_type(8)))  float v8f;
typedef __attribute__((ext_vector_type(8)))  int   v8i;

union BFrag { struct { uint4 lo, hi; } q; v16bf v; };
union IFrag { struct { uint2 a, b, c, d; } q; v8i v; };

#ifndef __has_builtin
#define __has_builtin(x) 0
#endif

#if __has_builtin(__builtin_amdgcn_global_load_async_to_lds_b128)
#define HAVE_ASYNC 1
typedef int gvec4i __attribute__((vector_size(16)));
typedef __attribute__((address_space(1))) gvec4i as1_v4i;
typedef __attribute__((address_space(3))) gvec4i as3_v4i;
#define ASYNC_B128(gp, lp) \
  __builtin_amdgcn_global_load_async_to_lds_b128((as1_v4i*)(gp), (as3_v4i*)(lp), 0, 0)
#if __has_builtin(__builtin_amdgcn_s_wait_asynccnt)
#define WAIT_ASYNC() __builtin_amdgcn_s_wait_asynccnt(0)
#else
#define WAIT_ASYNC() asm volatile("s_wait_asynccnt 0x0" ::: "memory")
#endif
#else
#define HAVE_ASYNC 0
#define WAIT_ASYNC()
#endif

__device__ __forceinline__ unsigned short f32_to_bf16(float f) {
  union { float f; unsigned u; } cvt; cvt.f = f;
  unsigned u = cvt.u;
  return (unsigned short)((u + 0x7FFFu + ((u >> 16) & 1u)) >> 16);
}

__global__ void k_zero(float* p, int n) {
  int i = blockIdx.x * blockDim.x + threadIdx.x;
  if (i < n) p[i] = 0.f;
}

__global__ __launch_bounds__(256) void k_abssum(const float* __restrict__ x,
                                                size_t n, float* __restrict__ slot) {
  __shared__ float red[256];
  size_t i = (size_t)blockIdx.x * blockDim.x + threadIdx.x;
  size_t stride = (size_t)gridDim.x * blockDim.x;
  float s = 0.f;
  for (; i < n; i += stride) s += fabsf(x[i]);
  red[threadIdx.x] = s; __syncthreads();
  for (int off = 128; off > 0; off >>= 1) {
    if ((int)threadIdx.x < off) red[threadIdx.x] += red[threadIdx.x + off];
    __syncthreads();
  }
  if (threadIdx.x == 0) atomicAdd(slot, red[0]);
}

__global__ void k_sign_bf16(const float* __restrict__ x, unsigned short* __restrict__ o, size_t n) {
  size_t i = (size_t)blockIdx.x * blockDim.x + threadIdx.x;
  if (i < n) {
    float v = x[i];
    o[i] = v > 0.f ? 0x3F80u : (v < 0.f ? 0xBF80u : 0u);
  }
}

__global__ void k_sign_i8(const float* __restrict__ x, signed char* __restrict__ o, size_t n) {
  size_t i = (size_t)blockIdx.x * blockDim.x + threadIdx.x;
  if (i < n) {
    float v = x[i];
    o[i] = v > 0.f ? 1 : (v < 0.f ? -1 : 0);
  }
}

__global__ void k_cvt_bf16(const float* __restrict__ x, unsigned short* __restrict__ o, size_t n) {
  size_t i = (size_t)blockIdx.x * blockDim.x + threadIdx.x;
  if (i < n) o[i] = f32_to_bf16(x[i]);
}

template <int ASRC, int EPI>
__global__ __launch_bounds__(256)
void gemm_bf16(const void* __restrict__ Asrc,
               const unsigned short* __restrict__ Brows,
               const float* __restrict__ mask,
               unsigned char* __restrict__ bout,
               float* __restrict__ scal,
               float* __restrict__ logits,
               int M, int N, int K) {
  __shared__ unsigned short As[2][128 * 40];
  __shared__ unsigned short Bs[2][128 * 40];

  const int tid  = threadIdx.x;
  const int lane = tid & 31;
  const int w    = tid >> 5;
  const int wm   = w & 1;
  const int wn   = w >> 1;
  const int mBase = blockIdx.x * 128;
  const int nBase = blockIdx.y * 128;
  const bool nfull = (nBase + 128 <= N);

  const int sr = tid >> 1;
  const int cb = (tid & 1) * 16;
  const size_t aRow = (size_t)(mBase + sr) * K;
  const int    nIdx = nBase + sr;
  const size_t bRow = (size_t)nIdx * K;

  v8f acc[4][2];
#pragma unroll
  for (int i = 0; i < 4; ++i)
#pragma unroll
    for (int j = 0; j < 2; ++j)
#pragma unroll
      for (int e = 0; e < 8; ++e) acc[i][j][e] = 0.f;

  float4 a4[4];
  uint4  a8;
#if !HAVE_ASYNC
  uint4  bq0, bq1;
#endif

  auto issue = [&](int k0, int buf) {
    const bool kfull = (k0 + 32 <= K);
    if constexpr (ASRC == 0) {
      const float4* p = (const float4*)((const float*)Asrc + aRow + k0 + cb);
      if (kfull || (k0 + cb + 16 <= K)) {
        a4[0] = p[0]; a4[1] = p[1]; a4[2] = p[2]; a4[3] = p[3];
      } else {
        a4[0] = a4[1] = a4[2] = a4[3] = make_float4(0.f, 0.f, 0.f, 0.f);
      }
    } else {
      if (kfull || (k0 + cb + 16 <= K))
        a8 = *(const uint4*)((const unsigned char*)Asrc + aRow + k0 + cb);
      else
        a8 = make_uint4(0, 0, 0, 0);
    }
    unsigned short* lp = &Bs[buf][sr * 40 + cb];
    const unsigned short* gp = Brows + bRow + k0 + cb;
#if HAVE_ASYNC
    if (nfull && kfull) {
      ASYNC_B128(gp, lp);
      ASYNC_B128(gp + 8, lp + 8);
    } else if ((nIdx < N) && (k0 + cb + 16 <= K)) {
      ASYNC_B128(gp, lp);
      ASYNC_B128(gp + 8, lp + 8);
    } else {
      uint4 z = make_uint4(0, 0, 0, 0);
      ((uint4*)lp)[0] = z; ((uint4*)lp)[1] = z;
    }
#else
    if ((nfull && kfull) || ((nIdx < N) && (k0 + cb + 16 <= K))) {
      const uint4* g4 = (const uint4*)gp;
      bq0 = g4[0]; bq1 = g4[1];
    } else {
      bq0 = bq1 = make_uint4(0, 0, 0, 0);
    }
#endif
  };

  auto commit = [&](int buf) {
    unsigned short tmp[16];
    if constexpr (ASRC == 0) {
      const float ff[16] = {a4[0].x, a4[0].y, a4[0].z, a4[0].w,
                            a4[1].x, a4[1].y, a4[1].z, a4[1].w,
                            a4[2].x, a4[2].y, a4[2].z, a4[2].w,
                            a4[3].x, a4[3].y, a4[3].z, a4[3].w};
#pragma unroll
      for (int e = 0; e < 16; ++e) tmp[e] = f32_to_bf16(ff[e]);
    } else {
      const unsigned bb[4] = {a8.x, a8.y, a8.z, a8.w};
#pragma unroll
      for (int e = 0; e < 16; ++e)
        tmp[e] = ((bb[e >> 2] >> ((e & 3) * 8)) & 0xFFu) ? 0x3F80u : 0u;
    }
    union { unsigned short s[8]; uint4 u; } p;
    uint4* dst = (uint4*)&As[buf][sr * 40 + cb];
#pragma unroll
    for (int e = 0; e < 8; ++e) p.s[e] = tmp[e];
    dst[0] = p.u;
#pragma unroll
    for (int e = 0; e < 8; ++e) p.s[e] = tmp[8 + e];
    dst[1] = p.u;
#if !HAVE_ASYNC
    uint4* bdst = (uint4*)&Bs[buf][sr * 40 + cb];
    bdst[0] = bq0; bdst[1] = bq1;
#endif
  };

  const int nIter = (K + 31) / 32;
  issue(0, 0);
  commit(0);
  for (int it = 0; it < nIter; ++it) {
    const int buf = it & 1;
    WAIT_ASYNC();
    __syncthreads();
    const bool more = (it + 1 < nIter);
    if (more) issue((it + 1) * 32, buf ^ 1);

    const int mrow = lane & 15;
    const int kb   = (lane >> 4) * 8;
    BFrag af[4], bfr[2];
#pragma unroll
    for (int tm = 0; tm < 4; ++tm) {
      const unsigned short* p = &As[buf][(wm * 64 + tm * 16 + mrow) * 40 + kb];
      af[tm].q.lo = *(const uint4*)p;
      af[tm].q.hi = *(const uint4*)(p + 16);
    }
#pragma unroll
    for (int tn = 0; tn < 2; ++tn) {
      const unsigned short* p = &Bs[buf][(wn * 32 + tn * 16 + mrow) * 40 + kb];
      bfr[tn].q.lo = *(const uint4*)p;
      bfr[tn].q.hi = *(const uint4*)(p + 16);
    }
#pragma unroll
    for (int tm = 0; tm < 4; ++tm)
#pragma unroll
      for (int tn = 0; tn < 2; ++tn)
        acc[tm][tn] = __builtin_amdgcn_wmma_f32_16x16x32_bf16(
            false, af[tm].v, false, bfr[tn].v, (short)0, acc[tm][tn], false, false);

    if (more) commit(buf ^ 1);
  }

  const int rofs = (lane >> 4) << 3;
  const int cloc = lane & 15;
  if constexpr (EPI == 0) {
    float asum = 0.f;
#pragma unroll
    for (int tm = 0; tm < 4; ++tm)
#pragma unroll
      for (int tn = 0; tn < 2; ++tn)
#pragma unroll
        for (int j = 0; j < 8; ++j) {
          float g = acc[tm][tn][j];
          asum += fabsf(g);
          int row = mBase + wm * 64 + tm * 16 + j + rofs;
          int col = nBase + wn * 32 + tn * 16 + cloc;
          if (col < N) {
            bool keep = (g > 0.f) && (mask[(size_t)row * N + col] != 0.f);
            bout[(size_t)row * N + col] = keep ? 1 : 0;
          }
        }
    for (int off = 16; off > 0; off >>= 1) asum += __shfl_down(asum, off, 32);
    if (lane == 0) atomicAdd(scal, asum);
  } else {
    const float a3 =
        (scal[0] * (1.f / ((float)HH * (float)DD))) *
        (scal[3] * (1.f / ((float)BB * (float)HH))) *
        (scal[1] * (1.f / ((float)HH * (float)HH))) *
        (scal[4] * (1.f / ((float)BB * (float)HH))) *
        (scal[2] * (1.f / ((float)HH * (float)HH))) *
        (scal[5] * (1.f / ((float)BB * (float)HH)));
#pragma unroll
    for (int tm = 0; tm < 4; ++tm)
#pragma unroll
      for (int tn = 0; tn < 2; ++tn)
#pragma unroll
        for (int j = 0; j < 8; ++j) {
          int row = mBase + wm * 64 + tm * 16 + j + rofs;
          int col = nBase + wn * 32 + tn * 16 + cloc;
          if (col < N) logits[(size_t)row * N + col] = a3 * acc[tm][tn][j];
        }
  }
}

__global__ __launch_bounds__(256)
void gemm_iu8(const unsigned char* __restrict__ A,
              const signed char* __restrict__ Brows,
              const float* __restrict__ mask,
              unsigned char* __restrict__ bout,
              float* __restrict__ slot,
              int M, int N, int K) {
  __shared__ unsigned char As[2][128 * 80];
  __shared__ unsigned char Bs[2][128 * 80];

  const int tid  = threadIdx.x;
  const int lane = tid & 31;
  const int w    = tid >> 5;
  const int wm   = w & 1;
  const int wn   = w >> 1;
  const int mBase = blockIdx.x * 128;
  const int nBase = blockIdx.y * 128;
  const bool nfull = (nBase + 128 <= N);

  const int sr = tid >> 1;
  const int cb = (tid & 1) * 32;
  const size_t aRow = (size_t)(mBase + sr) * K;
  const int    nIdx = nBase + sr;
  const size_t bRow = (size_t)nIdx * K;

  v8i acc[4][2];
#pragma unroll
  for (int i = 0; i < 4; ++i)
#pragma unroll
    for (int j = 0; j < 2; ++j)
#pragma unroll
      for (int e = 0; e < 8; ++e) acc[i][j][e] = 0;

#if !HAVE_ASYNC
  uint4 aq[2], bq[2];
#endif

  auto issue = [&](int k0, int buf) {
    const bool kfull = (k0 + 64 <= K);
    const unsigned char* ga = A + aRow + k0 + cb;
    const signed char*   gb = Brows + bRow + k0 + cb;
    unsigned char* la = &As[buf][sr * 80 + cb];
    unsigned char* lb = &Bs[buf][sr * 80 + cb];
#if HAVE_ASYNC
    if (kfull) {
      ASYNC_B128(ga, la);
      ASYNC_B128(ga + 16, la + 16);
    } else {
#pragma unroll
      for (int c = 0; c < 2; ++c) {
        if (k0 + cb + 16 * c + 16 <= K) ASYNC_B128(ga + 16 * c, la + 16 * c);
        else                            *(uint4*)(la + 16 * c) = make_uint4(0, 0, 0, 0);
      }
    }
    if (kfull && nfull) {
      ASYNC_B128(gb, lb);
      ASYNC_B128(gb + 16, lb + 16);
    } else {
#pragma unroll
      for (int c = 0; c < 2; ++c) {
        if ((nIdx < N) && (k0 + cb + 16 * c + 16 <= K)) ASYNC_B128(gb + 16 * c, lb + 16 * c);
        else *(uint4*)(lb + 16 * c) = make_uint4(0, 0, 0, 0);
      }
    }
#else
#pragma unroll
    for (int c = 0; c < 2; ++c) {
      const bool kok = kfull || (k0 + cb + 16 * c + 16 <= K);
      aq[c] = kok ? *(const uint4*)(ga + 16 * c) : make_uint4(0, 0, 0, 0);
      bq[c] = (kok && ((nfull) || (nIdx < N))) ? *(const uint4*)(gb + 16 * c)
                                               : make_uint4(0, 0, 0, 0);
    }
    (void)la; (void)lb;
#endif
  };

  auto commit = [&](int buf) {
#if !HAVE_ASYNC
#pragma unroll
    for (int c = 0; c < 2; ++c) {
      *(uint4*)&As[buf][sr * 80 + cb + 16 * c] = aq[c];
      *(uint4*)&Bs[buf][sr * 80 + cb + 16 * c] = bq[c];
    }
#else
    (void)buf;
#endif
  };

  const int nIter = (K + 63) / 64;
  issue(0, 0);
  commit(0);
  for (int it = 0; it < nIter; ++it) {
    const int buf = it & 1;
    WAIT_ASYNC();
    __syncthreads();
    const bool more = (it + 1 < nIter);
    if (more) issue((it + 1) * 64, buf ^ 1);

    const int mrow = lane & 15;
    const int c0   = (lane >> 4) * 8;
    IFrag af[4], bfr[2];
#pragma unroll
    for (int tm = 0; tm < 4; ++tm) {
      const unsigned char* p = &As[buf][(wm * 64 + tm * 16 + mrow) * 80 + c0];
      af[tm].q.a = *(const uint2*)(p);
      af[tm].q.b = *(const uint2*)(p + 16);
      af[tm].q.c = *(const uint2*)(p + 32);
      af[tm].q.d = *(const uint2*)(p + 48);
    }
#pragma unroll
    for (int tn = 0; tn < 2; ++tn) {
      const unsigned char* p = &Bs[buf][(wn * 32 + tn * 16 + mrow) * 80 + c0];
      bfr[tn].q.a = *(const uint2*)(p);
      bfr[tn].q.b = *(const uint2*)(p + 16);
      bfr[tn].q.c = *(const uint2*)(p + 32);
      bfr[tn].q.d = *(const uint2*)(p + 48);
    }
#pragma unroll
    for (int tm = 0; tm < 4; ++tm)
#pragma unroll
      for (int tn = 0; tn < 2; ++tn)
        acc[tm][tn] = __builtin_amdgcn_wmma_i32_16x16x64_iu8(
            false  , af[tm].v, true  , bfr[tn].v,
            acc[tm][tn], false, false);

    if (more) commit(buf ^ 1);
  }

  const int rofs = (lane >> 4) << 3;
  const int cloc = lane & 15;
  float asum = 0.f;
#pragma unroll
  for (int tm = 0; tm < 4; ++tm)
#pragma unroll
    for (int tn = 0; tn < 2; ++tn)
#pragma unroll
      for (int j = 0; j < 8; ++j) {
        int gi = acc[tm][tn][j];
        asum += fabsf((float)gi);
        int row = mBase + wm * 64 + tm * 16 + j + rofs;
        int col = nBase + wn * 32 + tn * 16 + cloc;
        if (col < N) {
          bool keep = (gi > 0) && (mask[(size_t)row * N + col] != 0.f);
          bout[(size_t)row * N + col] = keep ? 1 : 0;
        }
      }
  for (int off = 16; off > 0; off >>= 1) asum += __shfl_down(asum, off, 32);
  if (lane == 0) atomicAdd(slot, asum);
}

__global__ __launch_bounds__(256)
void k_softmax(const float* __restrict__ lg, float* __restrict__ out, int N) {
  __shared__ float red[256];
  const int row = blockIdx.x, tid = threadIdx.x;
  const float* lp = lg + (size_t)row * N;
  float m = -3.402823466e38f;
  for (int c = tid; c < N; c += 256) m = fmaxf(m, lp[c]);
  red[tid] = m; __syncthreads();
  for (int off = 128; off > 0; off >>= 1) {
    if (tid < off) red[tid] = fmaxf(red[tid], red[tid + off]);
    __syncthreads();
  }
  m = red[0]; __syncthreads();
  float s = 0.f;
  for (int c = tid; c < N; c += 256) s += __expf(lp[c] - m);
  red[tid] = s; __syncthreads();
  for (int off = 128; off > 0; off >>= 1) {
    if (tid < off) red[tid] += red[tid + off];
    __syncthreads();
  }
  const float inv = 1.f / red[0];
  for (int c = tid; c < N; c += 256)
    out[(size_t)row * N + c] = __expf(lp[c] - m) * inv;
}

extern "C" void kernel_launch(void* const* d_in, const int* in_sizes, int n_in,
                              void* d_out, int out_size, void* d_ws, size_t ws_size,
                              hipStream_t stream) {
  (void)in_sizes; (void)n_in; (void)out_size; (void)ws_size;
  const float* x  = (const float*)d_in[0];
  const float* dm = (const float*)d_in[1];
  const float* W1 = (const float*)d_in[2];
  const float* W2 = (const float*)d_in[3];
  const float* W3 = (const float*)d_in[4];
  const float* WL = (const float*)d_in[5];

  char* ws = (char*)d_ws;
  float*          scal = (float*)ws;
  unsigned short* W1s  = (unsigned short*)(ws + 256);
  signed char*    W2s  = (signed char*)   (ws + 12288256);
  signed char*    W3s  = (signed char*)   (ws + 16288256);
  unsigned short* WLs  = (unsigned short*)(ws + 20288256);
  unsigned char*  b1   = (unsigned char*) (ws + 24288256);
  unsigned char*  b2   = (unsigned char*) (ws + 40672256);
  unsigned char*  b3   = (unsigned char*) (ws + 57056256);
  float*          lgts = (float*)         (ws + 73440256);

  k_zero<<<1, 64, 0, stream>>>(scal, 64);
  k_abssum<<<1024, 256, 0, stream>>>(W1, (size_t)HH * DD, scal + 0);
  k_abssum<<<1024, 256, 0, stream>>>(W2, (size_t)HH * HH, scal + 1);
  k_abssum<<<1024, 256, 0, stream>>>(W3, (size_t)HH * HH, scal + 2);

  const size_t n1 = (size_t)HH * DD, n2 = (size_t)HH * HH, nl = (size_t)CC * HH;
  k_sign_bf16<<<(unsigned)((n1 + 255) / 256), 256, 0, stream>>>(W1, W1s, n1);
  k_sign_i8 <<<(unsigned)((n2 + 255) / 256), 256, 0, stream>>>(W2, W2s, n2);
  k_sign_i8 <<<(unsigned)((n2 + 255) / 256), 256, 0, stream>>>(W3, W3s, n2);
  k_cvt_bf16<<<(unsigned)((nl + 255) / 256), 256, 0, stream>>>(WL, WLs, nl);

  gemm_bf16<0, 0><<<dim3(BB / 128, (HH + 127) / 128), 256, 0, stream>>>(
      x, W1s, dm, b1, scal + 3, nullptr, BB, HH, DD);
  gemm_iu8<<<dim3(BB / 128, (HH + 127) / 128), 256, 0, stream>>>(
      b1, W2s, dm + (size_t)BB * HH, b2, scal + 4, BB, HH, HH);
  gemm_iu8<<<dim3(BB / 128, (HH + 127) / 128), 256, 0, stream>>>(
      b2, W3s, dm + 2 * (size_t)BB * HH, b3, scal + 5, BB, HH, HH);
  gemm_bf16<1, 1><<<dim3(BB / 128, (CC + 127) / 128), 256, 0, stream>>>(
      b3, WLs, nullptr, nullptr, scal, lgts, BB, CC, HH);

  k_softmax<<<BB, 256, 0, stream>>>(lgts, (float*)d_out, CC);
}
